// MultiHeadAttention_26001732010029
// MI455X (gfx1250) — matrix operand order repaired and checked by reading, not compiled, not run on hardware
//
#include <hip/hip_runtime.h>
#include <hip/hip_bf16.h>

typedef __attribute__((ext_vector_type(16))) _Float16 v16h;
typedef __attribute__((ext_vector_type(8)))  _Float16 v8h;
typedef __attribute__((ext_vector_type(8)))  float    v8f;
typedef __attribute__((ext_vector_type(4)))  float    v4f;

#define LDA 264   // row stride (halves) for [w][c] buffers (256 + pad, mult of 8)
#define LDV 136   // row stride for v [c][w] (128 + pad)
#define LDP 136   // row stride for probs stage [16][128 + pad]

// ---- fragment builders (ISA 7.12.2 wave32 layouts) ----
__device__ __forceinline__ v16h frag_cat(v8h a, v8h b) {
    return __builtin_shufflevector(a, b, 0,1,2,3,4,5,6,7,8,9,10,11,12,13,14,15);
}

// A matrix 16x32 f16: lane<16 row M=lane holds K {0..7,16..23}; lane>=16 holds K {8..15,24..31}
__device__ __forceinline__ v16h fragA(const _Float16* __restrict__ p0, int lda,
                                      int m0, int k0, int lane) {
    int hi = (lane >> 4) & 1;
    const _Float16* p = p0 + (size_t)(m0 + (lane & 15)) * lda + k0 + hi * 8;
    v8h a = *(const v8h*)p;
    v8h b = *(const v8h*)(p + 16);
    return frag_cat(a, b);
}

// B matrix 32x16 f16: lane%16 = column N; lane<16 holds K {0..7,16..23}; lane>=16 holds K {8..15,24..31}
// source stored [n][k] with k contiguous
__device__ __forceinline__ v16h fragB(const _Float16* __restrict__ p0, int lda,
                                      int n0, int k0, int lane) {
    int hi = (lane >> 4) & 1;
    const _Float16* p = p0 + (size_t)(n0 + (lane & 15)) * lda + k0 + hi * 8;
    v8h a = *(const v8h*)p;
    v8h b = *(const v8h*)(p + 16);
    return frag_cat(a, b);
}

__device__ __forceinline__ v8f wmma16(v16h a, v16h b, v8f c) {
    return __builtin_amdgcn_wmma_f32_16x16x32_f16(false, a, false, b, (short)0, c, false, false);
}

// ---- phase-1 projection GEMM: MODE 0=q (scaled), 1=k, 2=v ----
// q/k results go to [w][c] buffers with a single packed ds_store_b128 per tile;
// v goes to [c][w] with per-row b16 stores.
template <int MODE>
__device__ __forceinline__ void gemm_proj(const _Float16* __restrict__ Wg,
                                          const float* __restrict__ bias,
                                          const _Float16* __restrict__ Xh,
                                          _Float16* __restrict__ dstT,   // [w][c] (q/k)
                                          _Float16* __restrict__ dstV,   // [c][w] (v)
                                          int wave, int lane, int hi, int nL) {
    const float sscale = 0.17677669529663687f;   // 1/sqrt(32)
    #pragma unroll
    for (int mi = 0; mi < 2; ++mi) {
        int m0 = (wave * 2 + mi) * 16;
        v8f bias8 = *(const v8f*)(bias + m0 + hi * 8);
        for (int nt = 0; nt < 8; ++nt) {
            int n0 = nt * 16;
            v8f acc = {};
            #pragma unroll
            for (int k0 = 0; k0 < 256; k0 += 32)
                acc = wmma16(fragA(Wg, 256, m0, k0, lane),
                             fragB(Xh, LDA, n0, k0, lane), acc);
            acc += bias8;
            if (MODE == 0) acc *= sscale;
            if (MODE < 2) {
                v8h hc;
                #pragma unroll
                for (int r = 0; r < 8; ++r) hc[r] = (_Float16)acc[r];
                // lane's 8 rows are consecutive channels o = m0+8*hi .. +7 -> one b128 store
                *(v8h*)(dstT + (n0 + nL) * LDA + m0 + 8 * hi) = hc;
            } else {
                #pragma unroll
                for (int r = 0; r < 8; ++r)
                    dstV[(m0 + r + 8 * hi) * LDV + n0 + nL] = (_Float16)acc[r];
            }
        }
    }
}

// ---- prepass: convert the five 256x256 f32 weight matrices to f16 in d_ws ----
__global__ void pack_weights(const float* __restrict__ a, const float* __restrict__ b,
                             const float* __restrict__ c, const float* __restrict__ d,
                             const float* __restrict__ e, _Float16* __restrict__ o) {
    int i = blockIdx.x * 256 + threadIdx.x;     // 5 * 65536 total
    int w = i >> 16, j = i & 65535;
    const float* s = (w == 0) ? a : (w == 1) ? b : (w == 2) ? c : (w == 3) ? d : e;
    o[i] = (_Float16)s[j];
}

// ---- fused row-attention kernel: one workgroup per (b, h) image row ----
__global__ void __launch_bounds__(256, 1)
attn_row_kernel(const float* __restrict__ x,
                const _Float16* __restrict__ Wh,   // packed f16: Wq,Wk,Wv,Wfc,Wp (65536 each)
                const float* __restrict__ bq, const float* __restrict__ bk,
                const float* __restrict__ bv, const float* __restrict__ bfc,
                const float* __restrict__ bp,
                float* __restrict__ out) {
    extern __shared__ __align__(16) _Float16 smem[];
    _Float16* Xh = smem;                          // [128][LDA]  x row, [w][c]
    _Float16* qT = smem + 128 * LDA;              // [128][LDA]  q (later aliased by ctx)
    _Float16* kT = smem + 2 * 128 * LDA;          // [128][LDA]  k, [w][c]
    _Float16* vM = smem + 3 * 128 * LDA;          // [256][LDV]  v, [c][w]
    _Float16* ps = smem + 3 * 128 * LDA + 256 * LDV; // 8 waves x [16][LDP] probs stage

    const int tid  = threadIdx.x;
    const int wave = tid >> 5, lane = tid & 31;
    const int hi   = (lane >> 4) & 1, nL = lane & 15;
    const int blk  = blockIdx.x;
    const int b = blk >> 7, hrow = blk & 127;
    const float* xrow = x + (((size_t)b * 256) * 128 + hrow) * 128;

    // ---- phase 0: stage x row into LDS as f16 [w][c] (coalesced 16B global loads) ----
    #pragma unroll 4
    for (int it = 0; it < 32; ++it) {
        int c  = (tid >> 5) + it * 8;
        int w4 = (tid & 31) * 4;
        v4f v = *(const v4f*)(xrow + (size_t)c * 16384 + w4);
        Xh[(w4 + 0) * LDA + c] = (_Float16)v.x;
        Xh[(w4 + 1) * LDA + c] = (_Float16)v.y;
        Xh[(w4 + 2) * LDA + c] = (_Float16)v.z;
        Xh[(w4 + 3) * LDA + c] = (_Float16)v.w;
    }
    __syncthreads();

    // ---- phase 1: q/k/v = W*x + bias (q pre-scaled by 1/sqrt(32)) ----
    gemm_proj<0>(Wh,             bq, Xh, qT, nullptr, wave, lane, hi, nL);
    gemm_proj<1>(Wh +     65536, bk, Xh, kT, nullptr, wave, lane, hi, nL);
    gemm_proj<2>(Wh + 2 * 65536, bv, Xh, nullptr, vM, wave, lane, hi, nL);
    __syncthreads();

    // ---- phase 2: wave == head; flash-style row attention over width ----
    const int ch = wave * 32;                    // head's channel base
    _Float16* pw = ps + wave * 16 * LDP;

    // k B-fragments are invariant across q blocks: hoist all 8 (64 VGPRs)
    v16h kf[8];
    #pragma unroll
    for (int nb = 0; nb < 8; ++nb) kf[nb] = fragB(kT, LDA, nb * 16, ch, lane);

    for (int mb = 0; mb < 8; ++mb) {
        int m0 = mb * 16;
        v16h aq = fragA(qT, LDA, m0, ch, lane);  // q block [16 w][32 d]
        v8f sc[8];
        #pragma unroll
        for (int nb = 0; nb < 8; ++nb) {
            v8f z = {};
            sc[nb] = wmma16(aq, kf[nb], z);
        }
        // softmax: rows 0..7 live in lanes 0..15, rows 8..15 in lanes 16..31
        float rinv[8];
        #pragma unroll
        for (int r = 0; r < 8; ++r) {
            float m = sc[0][r];
            #pragma unroll
            for (int nb = 1; nb < 8; ++nb) m = fmaxf(m, sc[nb][r]);
            #pragma unroll
            for (int off = 1; off < 16; off <<= 1) m = fmaxf(m, __shfl_xor(m, off, 32));
            float s = 0.f;
            #pragma unroll
            for (int nb = 0; nb < 8; ++nb) {
                float e = __expf(sc[nb][r] - m);
                sc[nb][r] = e; s += e;
            }
            #pragma unroll
            for (int off = 1; off < 16; off <<= 1) s += __shfl_xor(s, off, 32);
            rinv[r] = 1.0f / s;
        }
        #pragma unroll
        for (int nb = 0; nb < 8; ++nb)
            #pragma unroll
            for (int r = 0; r < 8; ++r)
                pw[(r + 8 * hi) * LDP + nb * 16 + nL] = (_Float16)(sc[nb][r] * rinv[r]);

        // ctx = probs @ v  -> write into ctxT (aliases qT; q rows already consumed)
        #pragma unroll
        for (int db = 0; db < 2; ++db) {
            v8f accC = {};
            #pragma unroll
            for (int kb = 0; kb < 4; ++kb)
                accC = wmma16(fragA(pw, LDP, 0, kb * 32, lane),
                              fragB(vM, LDV, ch + db * 16, kb * 32, lane), accC);
            #pragma unroll
            for (int r = 0; r < 8; ++r)
                qT[(m0 + r + 8 * hi) * LDA + ch + db * 16 + nL] = (_Float16)accC[r];
        }
    }
    __syncthreads();

    // ---- phase 3: out = Wfc*ctx + Wp*x + (bfc + bp) ----
    const _Float16* Wfc_h = Wh + 3 * 65536;
    const _Float16* Wp_h  = Wh + 4 * 65536;
    #pragma unroll
    for (int mi = 0; mi < 2; ++mi) {
        int m0 = (wave * 2 + mi) * 16;
        v8f bias8 = *(const v8f*)(bfc + m0 + hi * 8);
        v8f biasp = *(const v8f*)(bp  + m0 + hi * 8);
        bias8 += biasp;
        for (int nt = 0; nt < 8; ++nt) {
            int n0 = nt * 16;
            v8f acc = bias8;
            #pragma unroll
            for (int k0 = 0; k0 < 256; k0 += 32)
                acc = wmma16(fragA(Wfc_h, 256, m0, k0, lane),
                             fragB(qT, LDA, n0, k0, lane), acc);   // qT holds ctx now
            #pragma unroll
            for (int k0 = 0; k0 < 256; k0 += 32)
                acc = wmma16(fragA(Wp_h, 256, m0, k0, lane),
                             fragB(Xh, LDA, n0, k0, lane), acc);
            #pragma unroll
            for (int r = 0; r < 8; ++r) {
                int o = m0 + r + 8 * hi;
                out[(((size_t)b * 256 + o) * 128 + hrow) * 128 + n0 + nL] = acc[r];
            }
        }
    }
}

extern "C" void kernel_launch(void* const* d_in, const int* in_sizes, int n_in,
                              void* d_out, int out_size, void* d_ws, size_t ws_size,
                              hipStream_t stream) {
    const float* x   = (const float*)d_in[0];
    const float* Wq  = (const float*)d_in[1];
    const float* bq  = (const float*)d_in[2];
    const float* Wk  = (const float*)d_in[3];
    const float* bk  = (const float*)d_in[4];
    const float* Wv  = (const float*)d_in[5];
    const float* bv  = (const float*)d_in[6];
    const float* Wfc = (const float*)d_in[7];
    const float* bfc = (const float*)d_in[8];
    const float* Wp  = (const float*)d_in[9];
    const float* bp  = (const float*)d_in[10];
    float* out = (float*)d_out;

    _Float16* Wh = (_Float16*)d_ws;                 // 5 * 65536 halves = 640 KB
    pack_weights<<<1280, 256, 0, stream>>>(Wq, Wk, Wv, Wfc, Wp, Wh);

    const size_t smem_bytes = (size_t)(3 * 128 * LDA + 256 * LDV + 8 * 16 * LDP) * 2; // 307200 B
    (void)hipFuncSetAttribute((const void*)attn_row_kernel,
                              hipFuncAttributeMaxDynamicSharedMemorySize,
                              (int)smem_bytes);
    attn_row_kernel<<<1024, 256, smem_bytes, stream>>>(x, Wh, bq, bk, bv, bfc, bp, out);
}
